// LSTMContextualize_69930657513539
// MI455X (gfx1250) — matrix operand order repaired and checked by reading, not compiled, not run on hardware
//
#include <hip/hip_runtime.h>
#include <hip/hip_bf16.h>
#include <math.h>

// Problem dims
#define BQ 16
#define TQ 512
#define EQ 1024
#define HQ 512
#define GN 1536          // 3*H gate columns
#define NWG 8            // workgroups per direction in recurrent phase
#define RBLK 384         // 12 waves: 3 gates x 4 col-tiles, 1 WMMA tile per wave

typedef __attribute__((ext_vector_type(16))) __bf16    v16bf;
typedef __attribute__((ext_vector_type(8)))  __bf16    v8bf;
typedef __attribute__((ext_vector_type(8)))  float     v8f;
typedef __attribute__((ext_vector_type(4)))  unsigned  v4u;
typedef __attribute__((ext_vector_type(4)))  int       v4i;
typedef __attribute__((ext_vector_type(8)))  int       v8i;

// ---- workspace layout (bytes) ----
static constexpr size_t OFF_XBF  = 0;                               // x as bf16 [B,T,E]
static constexpr size_t SZ_XBF   = (size_t)BQ * TQ * EQ * 2;
static constexpr size_t OFF_WX   = OFF_XBF + SZ_XBF;                // [2][GN][E] bf16
static constexpr size_t SZ_WX    = (size_t)2 * GN * EQ * 2;
static constexpr size_t OFF_WH   = OFF_WX + SZ_WX;                  // [2][GN][H] bf16
static constexpr size_t SZ_WH    = (size_t)2 * GN * HQ * 2;
static constexpr size_t OFF_GIN  = OFF_WH + SZ_WH;                  // [2][T][B][GN] f32
static constexpr size_t SZ_GIN   = (size_t)2 * TQ * BQ * GN * 4;
static constexpr size_t OFF_HBUF = OFF_GIN + SZ_GIN;                // [2buf][2dir][B][H] bf16
static constexpr size_t SZ_HBUF  = (size_t)2 * 2 * BQ * HQ * 2;
static constexpr size_t OFF_CTR  = OFF_HBUF + SZ_HBUF;              // 2 barrier counters

// ============================================================================
// Tensor Data Mover: 2D tile load Global -> LDS.
// D# per CDNA5 ISA ch.8: group0 = {count=1, lds_addr, global_addr(57b), type=2},
// group1 = {data_size=8B, tensor_dim0/1, tile_dim0/1, tensor_dim0_stride}.
// All dims/strides in 8-byte units. One wave issues; EXEC is ignored by TDM.
// ============================================================================
__device__ __forceinline__ void tdm_load_2d(unsigned lds_addr, const void* gaddr,
                                            unsigned tile_d0, unsigned tile_d1,
                                            unsigned tensor_d0, unsigned tensor_d1,
                                            unsigned stride0)
{
    unsigned long long ga = (unsigned long long)gaddr;
    v4u g0;
    g0.x = 1u;                                            // count=1 (valid), user mode
    g0.y = lds_addr;                                      // LDS byte address
    g0.z = (unsigned)ga;                                  // global_addr[31:0]
    g0.w = (unsigned)((ga >> 32) & 0x01FFFFFFu)           // global_addr[56:32]
         | (2u << 30);                                    // type = 2 ("image")
    v8i g1;
    g1[0] = (int)(3u << 16);                              // data_size = 3 (8 bytes)
    g1[1] = (int)((tensor_d0 & 0xFFFFu) << 16);                             // dim0 lo16
    g1[2] = (int)(((tensor_d0 >> 16) & 0xFFFFu) | ((tensor_d1 & 0xFFFFu) << 16));
    g1[3] = (int)(((tensor_d1 >> 16) & 0xFFFFu) | ((tile_d0 & 0xFFFFu) << 16));
    g1[4] = (int)(tile_d1 & 0xFFFFu);                     // tile_dim1 | tile_dim2=0
    g1[5] = (int)stride0;                                 // tensor_dim0_stride[31:0]
    g1[6] = 0;
    g1[7] = 0;
    v4i gz = {0, 0, 0, 0};
#if defined(__clang_major__) && (__clang_major__ >= 23)
    v8i gz8 = {0, 0, 0, 0, 0, 0, 0, 0};
    __builtin_amdgcn_tensor_load_to_lds(g0, g1, gz, gz, gz8, 0);
#else
    __builtin_amdgcn_tensor_load_to_lds(g0, g1, gz, gz, 0);
#endif
}

// ============================================================================
// Phase 0: fp32 -> bf16 conversions, weight split, initial h broadcast,
//          barrier-counter reset (keeps graph replay deterministic).
// ============================================================================
__global__ void lstm_convert_kernel(const float* __restrict__ x,
                                    const float* __restrict__ w_fwd,
                                    const float* __restrict__ w_bwd,
                                    const float* __restrict__ h0f,
                                    const float* __restrict__ h0b,
                                    char* __restrict__ ws)
{
    __bf16*   xbf  = (__bf16*)(ws + OFF_XBF);
    __bf16*   wx   = (__bf16*)(ws + OFF_WX);
    __bf16*   wh   = (__bf16*)(ws + OFF_WH);
    __bf16*   hbuf = (__bf16*)(ws + OFF_HBUF);
    unsigned* ctr  = (unsigned*)(ws + OFF_CTR);

    size_t tid = (size_t)blockIdx.x * blockDim.x + threadIdx.x;
    size_t nth = (size_t)gridDim.x * blockDim.x;

    for (size_t i = tid; i < (size_t)BQ * TQ * EQ; i += nth)
        xbf[i] = (__bf16)x[i];

    // Split W[:, :E] -> Wx, W[:, E:] -> Wh, both row-major bf16, per direction.
    for (size_t i = tid; i < (size_t)GN * (EQ + HQ); i += nth) {
        size_t r = i / (EQ + HQ), c = i % (EQ + HQ);
        float vf = w_fwd[i], vb = w_bwd[i];
        if (c < EQ) {
            wx[(size_t)0 * GN * EQ + r * EQ + c] = (__bf16)vf;
            wx[(size_t)1 * GN * EQ + r * EQ + c] = (__bf16)vb;
        } else {
            size_t cc = c - EQ;
            wh[(size_t)0 * GN * HQ + r * HQ + cc] = (__bf16)vf;
            wh[(size_t)1 * GN * HQ + r * HQ + cc] = (__bf16)vb;
        }
    }

    // hbuf[buf=0][d][b][col] = h0_d[col] (broadcast over batch)
    for (size_t i = tid; i < (size_t)2 * BQ * HQ; i += nth) {
        size_t d   = i / (BQ * HQ);
        size_t col = i % HQ;
        float v = (d == 0) ? h0f[col] : h0b[col];
        hbuf[(size_t)(0 * 2 + d) * BQ * HQ + (i % (BQ * HQ))] = (__bf16)v;
    }
    if (tid < 2) ctr[tid] = 0u;
}

// ============================================================================
// Phase 1: G_in[d][t][b][n] = bias_d[n] + sum_e x[b][t][e] * Wx_d[n][e]
// One WMMA tile per wave: M = batch(16), N = 16 gate cols, K = E = 1024.
// The x slab for timestep t (16 rows x 1024, 32 KB) is shared by all 8 waves:
// TDM-staged into LDS once (strided 2D tile), then read as A fragments.
// Grid: (12 n-tile groups of 8 waves, T, 2 dirs), block 256 = 8 waves.
// ============================================================================
__global__ void __launch_bounds__(256)
lstm_pregemm_kernel(const float* __restrict__ b_fwd,
                    const float* __restrict__ b_bwd,
                    char* __restrict__ ws)
{
    __shared__ __bf16 a_lds[BQ * EQ];                 // 32 KB: x[t] slab, row-major

    const int lane = threadIdx.x & 31;
    const int wave = threadIdx.x >> 5;
    const int ln   = lane & 15;       // A row (M) / B column (N) / C column
    const int sel  = lane >> 4;       // K-half selector: lane half takes k = 16*sel + s (contiguous)

    const int t = blockIdx.y;
    const int d = blockIdx.z;
    const int n0 = (blockIdx.x * 8 + wave) * 16;      // gate-column base, 0..1520

    const __bf16* xbf  = (const __bf16*)(ws + OFF_XBF);
    const __bf16* wx   = (const __bf16*)(ws + OFF_WX) + (size_t)d * GN * EQ;
    float*        gin  = (float*)(ws + OFF_GIN) + (size_t)(d * TQ + t) * BQ * GN;
    const float*  bias = (d == 0) ? b_fwd : b_bwd;

    // Stage x slab: rows b=0..15 at x[(b*T + t)*E], row stride T*E bf16.
    if (wave == 0) {
        tdm_load_2d((unsigned)(unsigned long long)(uintptr_t)a_lds,
                    xbf + (size_t)t * EQ,
                    /*tile_d0=*/EQ * 2 / 8, /*tile_d1=*/BQ,
                    /*tensor_d0=*/(unsigned)TQ * EQ * 2 / 8, /*tensor_d1=*/BQ,
                    /*stride0=*/(unsigned)TQ * EQ * 2 / 8);
        __builtin_amdgcn_s_wait_tensorcnt(0);
    }
    __syncthreads();

    const __bf16* arow = a_lds + (size_t)ln * EQ;     // x[b=ln][t][:]
    const __bf16* brow = wx + (size_t)(n0 + ln) * EQ; // Wx row = B column

    v8f acc = {};
    #pragma unroll 4
    for (int k0 = 0; k0 < EQ; k0 += 32) {
        v16bf a, b;
        a = *(const v16bf*)(arow + k0 + sel * 16);
        b = *(const v16bf*)(brow + k0 + sel * 16);
        acc = __builtin_amdgcn_wmma_f32_16x16x32_bf16(
                  false, a, false, b, (short)0, acc, false, false);
    }

    const int   nn = n0 + ln;
    const float bb = bias[nn];
    #pragma unroll
    for (int r = 0; r < 8; r++) {
        int m = r + sel * 8;                          // C/D: VGPR r, lane-half -> row
        gin[(size_t)m * GN + nn] = acc[r] + bb;
    }
}

// ============================================================================
// Phase 2: persistent recurrence. Per direction, 8 workgroups each own 64
// h-columns (12 waves x one 16x16 gate tile, K = H = 512).
// Wh fragments are loop-invariant -> preloaded into 16 v16bf registers per
// wave; per step only h (16 KB) moves: TDM global->LDS, then 16 back-to-back
// WMMAs. Double-buffered global h + per-step device barrier republishes h.
// ============================================================================
__global__ void __launch_bounds__(RBLK)
lstm_recurrent_kernel(const float* __restrict__ c0f,
                      const float* __restrict__ c0b,
                      float* __restrict__ out,
                      char* __restrict__ ws)
{
    __shared__ __bf16 h_lds[BQ * HQ];            // 16 KB: full h, shared by all waves
    __shared__ float  gate_lds[3][BQ][64];       // 12 KB: pre-activation gates
    __shared__ float  c_lds[BQ * 64];            //  4 KB: persistent cell state

    const int tid  = threadIdx.x;
    const int lane = tid & 31;
    const int wave = tid >> 5;        // 0..11
    const int g    = wave >> 2;       // gate: 0=i, 1=j, 2=o
    const int ct   = wave & 3;        // 16-col tile within our 64 columns
    const int ln   = lane & 15;
    const int sel  = lane >> 4;

    const int wg   = blockIdx.x;      // 0..NWG-1
    const int d    = blockIdx.y;      // direction
    const int col0 = wg * 64;         // owned h-column base

    const __bf16* wh      = (const __bf16*)(ws + OFF_WH) + (size_t)d * GN * HQ;
    const float*  ginbase = (const float*)(ws + OFF_GIN) + (size_t)d * TQ * BQ * GN;
    __bf16*       hbuf    = (__bf16*)(ws + OFF_HBUF);
    unsigned*     ctr     = (unsigned*)(ws + OFF_CTR) + d;
    const float*  c0      = (d == 0) ? c0f : c0b;

    for (int e = tid; e < BQ * 64; e += RBLK)
        c_lds[e] = c0[col0 + (e & 63)];

    const int n0g = g * HQ + col0 + ct * 16;           // global gate column base
    const __bf16* brow = wh + (size_t)(n0g + ln) * HQ; // Wh row = B column

    // Hoist the wave's entire Wh tile (16x512) into registers: 16 x v16bf.
    v16bf bfr[16];
    #pragma unroll
    for (int kk = 0; kk < 16; kk++)
        bfr[kk] = *(const v16bf*)(brow + kk * 32 + sel * 16);

    int cur = 0;
    for (int s = 0; s < TQ; s++) {
        const int tx = (d == 0) ? s : (TQ - 1 - s);

        // TDM-stage h[cur] (16 KB, contiguous) into LDS; one wave issues.
        if (wave == 0) {
            tdm_load_2d((unsigned)(unsigned long long)(uintptr_t)h_lds,
                        hbuf + (size_t)(cur * 2 + d) * BQ * HQ,
                        /*tile_d0=*/BQ * HQ * 2 / 8, /*tile_d1=*/1,
                        /*tensor_d0=*/BQ * HQ * 2 / 8, /*tensor_d1=*/1,
                        /*stride0=*/BQ * HQ * 2 / 8);
            __builtin_amdgcn_s_wait_tensorcnt(0);
        }
        __syncthreads();

        // 16x16 gate tile, K = 512: 16 back-to-back WMMAs, B from registers.
        v8f acc = {};
        const __bf16* arow = h_lds + (size_t)ln * HQ;
        #pragma unroll
        for (int kk = 0; kk < 16; kk++) {
            const int k0 = kk * 32;
            v16bf a = *(const v16bf*)(arow + k0 + sel * 16);
            acc = __builtin_amdgcn_wmma_f32_16x16x32_bf16(
                      false, a, false, bfr[kk], (short)0, acc, false, false);
        }

        // add precomputed input contribution, stash pre-activations
        const float* gin = ginbase + (size_t)tx * BQ * GN + n0g + ln;
        #pragma unroll
        for (int r = 0; r < 8; r++) {
            int m = r + sel * 8;
            gate_lds[g][m][ct * 16 + ln] = acc[r] + gin[(size_t)m * GN];
        }
        __syncthreads();

        // coupled-gate update + emit h (f32 to out, bf16 to next h buffer)
        __bf16* hnext = hbuf + (size_t)(((cur ^ 1) * 2 + d)) * BQ * HQ;
        for (int e = tid; e < BQ * 64; e += RBLK) {
            int m = e >> 6, cc = e & 63;
            float gi = gate_lds[0][m][cc];
            float gj = gate_lds[1][m][cc];
            float go = gate_lds[2][m][cc];
            float i_ = 1.f / (1.f + __expf(-gi));
            float c  = (1.f - i_) * c_lds[e] + i_ * tanhf(gj);
            float h  = tanhf(c) * (1.f / (1.f + __expf(-go)));
            c_lds[e] = c;
            hnext[(size_t)m * HQ + col0 + cc] = (__bf16)h;
            out[((size_t)m * TQ + tx) * (2 * HQ) + (size_t)d * HQ + col0 + cc] = h;
        }
        __syncthreads();

        // per-direction device barrier: monotonic arrival counter
        if (tid == 0) {
            __threadfence();
            atomicAdd(ctr, 1u);
            unsigned target = (unsigned)(NWG * (s + 1));
            while (__hip_atomic_load(ctr, __ATOMIC_RELAXED, __HIP_MEMORY_SCOPE_AGENT) < target) {
                __builtin_amdgcn_s_sleep(1);
            }
        }
        __syncthreads();
        __threadfence();
        cur ^= 1;
    }
}

// ============================================================================
extern "C" void kernel_launch(void* const* d_in, const int* in_sizes, int n_in,
                              void* d_out, int out_size, void* d_ws, size_t ws_size,
                              hipStream_t stream) {
    const float* x     = (const float*)d_in[0];
    const float* w_fwd = (const float*)d_in[1];
    const float* b_fwd = (const float*)d_in[2];
    const float* w_bwd = (const float*)d_in[3];
    const float* b_bwd = (const float*)d_in[4];
    const float* h0f   = (const float*)d_in[5];
    const float* h0b   = (const float*)d_in[6];
    const float* c0f   = (const float*)d_in[7];
    const float* c0b   = (const float*)d_in[8];
    float* out = (float*)d_out;
    char*  ws  = (char*)d_ws;

    lstm_convert_kernel<<<dim3(2048), dim3(256), 0, stream>>>(x, w_fwd, w_bwd, h0f, h0b, ws);
    lstm_pregemm_kernel<<<dim3(12, TQ, 2), dim3(256), 0, stream>>>(b_fwd, b_bwd, ws);
    lstm_recurrent_kernel<<<dim3(NWG, 2), dim3(RBLK), 0, stream>>>(c0f, c0b, out, ws);
}
